// SAE_70523363000341
// MI455X (gfx1250) — compile-verified, matrix operand order checked by reading, not run on hardware
//
#include <hip/hip_runtime.h>
#include <hip/hip_bf16.h>
#include <math.h>

// ---------------- problem constants ----------------
#define BB 8192
#define DM 1024
#define NF 32768
#define TOPK 32
#define AUXK 256
// TGT_NORM = sqrt(1024) = 32 exactly

// ---------------- output layout (fp32, concatenated flat in return order) ---
constexpr size_t OXN = 0;                         // x_normed  8192*1024
constexpr size_t OW  = OXN + (size_t)BB*DM;       // weights   8192*32
constexpr size_t OI  = OW  + (size_t)BB*TOPK;     // indices   8192*32
constexpr size_t OYN = OI  + (size_t)BB*TOPK;     // y_normed  8192*1024
constexpr size_t OY  = OYN + (size_t)BB*DM;       // y         8192*1024
constexpr size_t ORL = OY  + (size_t)BB*DM;       // recon_loss 8192
constexpr size_t OAL = ORL + BB;                  // aux_k_loss 8192
constexpr size_t OLS = OAL + BB;                  // loss       8192
constexpr size_t OFVU= OLS + BB;                  // fvu        1
constexpr size_t OVE = OFVU + 1;                  // var_explained 1

// ---------------- workspace layout (bytes) ----------------
constexpr size_t WS_A      = 0;                        // bf16 A (8192x1024)        16 MB
constexpr size_t WS_WT     = WS_A   + (size_t)BB*DM*2; // bf16 W_enc^T (32768x1024) 64 MB
constexpr size_t WS_DIDX   = WS_WT  + (size_t)NF*DM*2; // int dead idx[256]
constexpr size_t WS_DACT   = WS_DIDX + 1024;           // int dead act[256]
constexpr size_t WS_BD     = WS_DACT + 1024;           // bf16 WencDead^T (256x1024) 0.5 MB
constexpr size_t WS_BT     = WS_BD  + (size_t)AUXK*DM*2; // bf16 WdecDead^T (1024x256) 0.5 MB
constexpr size_t WS_DWB    = WS_BT  + (size_t)DM*AUXK*2; // bf16 dead weights (8192x256) 4 MB
constexpr size_t WS_AUXSQ  = WS_DWB + (size_t)BB*AUXK*2; // f32 aux sq per row [8192]
constexpr size_t WS_CS     = WS_AUXSQ + (size_t)BB*4;    // f32 col stats [1024*5]

// ---------------- WMMA types ----------------
typedef __bf16 v16bf __attribute__((ext_vector_type(16)));
typedef float  v8f   __attribute__((ext_vector_type(8)));
union FragU { v16bf v; uint4 q[2]; };

__device__ __forceinline__ unsigned short f2bf(float f) {
    union { float f; unsigned int u; } c; c.f = f;
    unsigned int u = c.u;
    unsigned int r = (u + 0x7fffu + ((u >> 16) & 1u)) >> 16;
    return (unsigned short)r;
}

__device__ __forceinline__ float block_reduce(float v, float* red, int tid, int n) {
    red[tid] = v; __syncthreads();
    for (int s = n >> 1; s > 0; s >>= 1) {
        if (tid < s) red[tid] += red[tid + s];
        __syncthreads();
    }
    float r = red[0]; __syncthreads();
    return r;
}

// ============ 1. normalize x, build bf16 pre-activation A ============
__global__ __launch_bounds__(256) void k_prep(const float* __restrict__ x,
                                              const float* __restrict__ b_pre,
                                              const float* __restrict__ b_post,
                                              const float* __restrict__ avg_norm,
                                              float* __restrict__ out,
                                              unsigned short* __restrict__ A) {
    size_t i = (size_t)blockIdx.x * 256 + threadIdx.x;   // over B*D
    float scale = 32.0f / avg_norm[0];
    int d = (int)(i & (DM - 1));
    float v = x[i] * scale;
    v = fminf(fmaxf(v, -16.0f), 16.0f);
    out[OXN + i] = v;
    A[i] = f2bf(v - b_post[d] - b_pre[d]);
}

// ============ 2. transpose W_enc (K x N) -> Wt (N x K) bf16 ============
__global__ __launch_bounds__(256) void k_wenc_t(const float* __restrict__ W,
                                                unsigned short* __restrict__ Wt) {
    __shared__ float t[32][33];
    int nb = blockIdx.x * 32, kb = blockIdx.y * 32;
    int tx = threadIdx.x, ty = threadIdx.y;              // (32,8)
    for (int j = ty; j < 32; j += 8)
        t[j][tx] = W[(size_t)(kb + j) * NF + nb + tx];
    __syncthreads();
    for (int j = ty; j < 32; j += 8)
        Wt[(size_t)(nb + j) * DM + kb + tx] = f2bf(t[tx][j]);
}

// ============ 3. fused encoder GEMM + top-32 + sparse decode ============
// 4 waves / block, 64 rows / block. Each wave keeps its 16-row A strip in
// 256 VGPRs and sweeps the FULL feature range (shared B addresses -> WGP$/L2
// reuse, no cross-wave merge). B fragments go through an 8-deep rotating
// register ring with prefetch distance 8 (spilling into the next N-tile) so
// the load pipe stays ~14 deep instead of serializing on s_wait_loadcnt 0.
__global__ __launch_bounds__(128) void k_main(const unsigned short* __restrict__ A,
                                              const unsigned short* __restrict__ Wt,
                                              const float* __restrict__ Wdec,
                                              const float* __restrict__ b_post,
                                              const float* __restrict__ avg_norm,
                                              float* __restrict__ out) {
    __shared__ float sTile[4 * 256];           // per-wave 16x16 fp32 tile
    __shared__ float sLV[4 * 16 * 32];         // per-(wave,row) top-32 values
    __shared__ int   sLI[4 * 16 * 32];         // per-(wave,row) top-32 indices
    __shared__ float sSelV[64 * 32];
    __shared__ int   sSelI[64 * 32];
    __shared__ float sRecon[64];

    int tid  = threadIdx.x;
    int wave = tid >> 5;
    int lane = tid & 31;
    int m16  = lane & 15;
    int h    = lane >> 4;
    int rowBase = blockIdx.x * 64;             // 64 rows per block
    int myRow   = rowBase + wave * 16 + m16;   // this lane's A row

    // ---- preload this wave's 16x1024 A strip into registers (256 VGPRs) ----
    FragU af[32];
    {
        const unsigned short* aRow = A + (size_t)myRow * DM;
        #pragma unroll
        for (int k = 0; k < 32; ++k) {
            af[k].q[0] = *(const uint4*)(aRow + k * 32 + 8 * h);
            af[k].q[1] = *(const uint4*)(aRow + k * 32 + 16 + 8 * h);
        }
    }

    if (tid < 64) sRecon[tid] = 0.0f;
    if (lane < 16) {
        int base = (wave * 16 + lane) * 32;
        for (int k = 0; k < 32; ++k) { sLV[base + k] = -INFINITY; sLI[base + k] = 0; }
    }
    __syncthreads();

    float cmin = -INFINITY; int cam = 0;

    // ---- prime the 8-deep B-fragment ring for the first tile ----
    FragU bb[8];
    {
        const unsigned short* bRow0 = Wt + (size_t)m16 * DM;
        #pragma unroll
        for (int p = 0; p < 8; ++p) {
            bb[p].q[0] = *(const uint4*)(bRow0 + p * 32 + 8 * h);
            bb[p].q[1] = *(const uint4*)(bRow0 + p * 32 + 16 + 8 * h);
        }
    }

    for (int nb = 0; nb < NF; nb += 16) {      // all waves sweep the same columns
        const unsigned short* bRowCur = Wt + (size_t)(nb + m16) * DM;
        const unsigned short* bRowNxt = bRowCur + (size_t)16 * DM; // next tile
        v8f acc = {0.f, 0.f, 0.f, 0.f, 0.f, 0.f, 0.f, 0.f};
        #pragma unroll
        for (int kk = 0; kk < 32; ++kk) {
            acc = __builtin_amdgcn_wmma_f32_16x16x32_bf16(false, af[kk].v, false,
                                                          bb[kk & 7].v,
                                                          (short)0, acc, false, false);
            // prefetch fragment (kk+8); past kk=23 this primes the NEXT tile
            const unsigned short* src = (kk + 8 < 32)
                ? bRowCur + (kk + 8) * 32
                : bRowNxt + (kk - 24) * 32;
            bb[kk & 7].q[0] = *(const uint4*)(src + 8 * h);
            bb[kk & 7].q[1] = *(const uint4*)(src + 16 + 8 * h);
        }
        // dump 16x16 tile: vgpr r holds row (r + 8h), lane m16 holds col m16
        float* tile = sTile + wave * 256;
        #pragma unroll
        for (int r = 0; r < 8; ++r) tile[(r + 8 * h) * 16 + m16] = acc[r];
        // streaming top-32: lanes 0-15 each own one row (in-order LDS within wave)
        if (lane < 16) {
            int base = (wave * 16 + lane) * 32;
            const float* trow = sTile + wave * 256 + lane * 16;
            for (int c = 0; c < 16; ++c) {
                float v = trow[c];
                if (v > cmin) {
                    sLV[base + cam] = v; sLI[base + cam] = nb + c;
                    float mn = sLV[base]; int am = 0;
                    for (int k = 1; k < 32; ++k) {
                        float tv = sLV[base + k];
                        if (tv < mn) { mn = tv; am = k; }
                    }
                    cmin = mn; cam = am;
                }
            }
        }
    }
    __syncthreads();

    // per-wave final descending sort of its own rows (no cross-wave merge)
    if (lane < 16) {
        int base = (wave * 16 + lane) * 32;
        int mrow = wave * 16 + lane;
        size_t row = (size_t)(rowBase + mrow);
        for (int j = 0; j < 32; ++j) {
            float best = -INFINITY; int bi = 0, bk = 0;
            for (int k = 0; k < 32; ++k) {
                float v = sLV[base + k];
                if (v > best) { best = v; bi = sLI[base + k]; bk = k; }
            }
            sLV[base + bk] = -INFINITY;
            sSelV[mrow * 32 + j] = best;
            sSelI[mrow * 32 + j] = bi;
            out[OW + row * 32 + j] = best;
            out[OI + row * 32 + j] = (float)bi;
        }
    }
    __syncthreads();

    // fused sparse decode + y_normed, y, recon_loss (128 threads x 64 rows)
    float yscale = avg_norm[0] * (1.0f / 32.0f);
    for (int m = 0; m < 64; ++m) {
        size_t row = (size_t)(rowBase + m);
        float acc8[8];
        #pragma unroll
        for (int j = 0; j < 8; ++j) acc8[j] = 0.0f;
        for (int k = 0; k < 32; ++k) {
            float w = sSelV[m * 32 + k];
            const float* wd = Wdec + (size_t)sSelI[m * 32 + k] * DM;
            #pragma unroll
            for (int j = 0; j < 8; ++j) acc8[j] += w * wd[tid + 128 * j];
        }
        float sq = 0.0f;
        #pragma unroll
        for (int j = 0; j < 8; ++j) {
            int c = tid + 128 * j;
            float yn = acc8[j] + b_post[c];
            out[OYN + row * DM + c] = yn;
            out[OY  + row * DM + c] = yn * yscale;
            float d = out[OXN + row * DM + c] - yn;
            sq += d * d;
        }
        atomicAdd(&sRecon[m], sq);
    }
    __syncthreads();
    if (tid < 64) out[ORL + rowBase + tid] = sRecon[tid] * (1.0f / 1024.0f);
}

// ============ 4. exact top-256 of activated_in ============
__global__ __launch_bounds__(1024) void k_dead_topk(const int* __restrict__ act,
                                                    int* __restrict__ dIdx,
                                                    int* __restrict__ dAct) {
    __shared__ unsigned char mask[NF];
    __shared__ unsigned long long sKey[1024];
    int tid = threadIdx.x;
    for (int i = tid; i < NF; i += 1024) mask[i] = 0;
    __syncthreads();
    for (int r = 0; r < AUXK; ++r) {
        unsigned long long best = 0ull;
        for (int i = tid; i < NF; i += 1024) {
            if (!mask[i]) {
                unsigned int av = (unsigned int)act[i] ^ 0x80000000u;
                unsigned long long key =
                    ((unsigned long long)av << 32) | (unsigned int)(~i);
                if (key > best) best = key;
            }
        }
        sKey[tid] = best; __syncthreads();
        for (int s = 512; s > 0; s >>= 1) {
            if (tid < s && sKey[tid + s] > sKey[tid]) sKey[tid] = sKey[tid + s];
            __syncthreads();
        }
        if (tid == 0) {
            unsigned long long k = sKey[0];
            int idx = (int)(~(unsigned int)k);
            dIdx[r] = idx;
            dAct[r] = (int)((unsigned int)(k >> 32) ^ 0x80000000u);
            mask[idx] = 1;
        }
        __syncthreads();
    }
}

// ============ 5. pack dead weight slices to bf16 (K-contiguous) ============
__global__ __launch_bounds__(256) void k_pack_enc(const float* __restrict__ Wenc,
                                                  const int* __restrict__ dIdx,
                                                  unsigned short* __restrict__ Bd) {
    int j = blockIdx.x; int idx = dIdx[j];
    for (int k = threadIdx.x; k < DM; k += 256)
        Bd[(size_t)j * DM + k] = f2bf(Wenc[(size_t)k * NF + idx]);
}
__global__ __launch_bounds__(256) void k_pack_dec(const float* __restrict__ Wdec,
                                                  const int* __restrict__ dIdx,
                                                  unsigned short* __restrict__ Bt) {
    int j = blockIdx.x; int idx = dIdx[j];
    for (int d = threadIdx.x; d < DM; d += 256)
        Bt[(size_t)d * AUXK + j] = f2bf(Wdec[(size_t)idx * DM + d]);
}

__global__ void k_zero(float* p, int n) {
    int i = blockIdx.x * blockDim.x + threadIdx.x;
    if (i < n) p[i] = 0.0f;
}

// ============ 6. aux GEMM 1: masked dead weights = A @ WencDead ============
__global__ __launch_bounds__(256) void k_aux_enc(const unsigned short* __restrict__ A,
                                                 const unsigned short* __restrict__ Bd,
                                                 const int* __restrict__ dAct,
                                                 unsigned short* __restrict__ dWb) {
    int wave = threadIdx.x >> 5, lane = threadIdx.x & 31;
    int g = blockIdx.x * 8 + wave;            // 8192 tiles = (512 mt) x (16 nt)
    int mt = g >> 4, nt = g & 15;
    int m16 = lane & 15, h = lane >> 4;
    const unsigned short* aRow = A + (size_t)(mt * 16 + m16) * DM;
    const unsigned short* bRow = Bd + (size_t)(nt * 16 + m16) * DM;
    v8f acc = {0.f, 0.f, 0.f, 0.f, 0.f, 0.f, 0.f, 0.f};
    for (int kk = 0; kk < DM; kk += 32) {
        FragU a, b;
        a.q[0] = *(const uint4*)(aRow + kk + 8 * h);
        a.q[1] = *(const uint4*)(aRow + kk + 16 + 8 * h);
        b.q[0] = *(const uint4*)(bRow + kk + 8 * h);
        b.q[1] = *(const uint4*)(bRow + kk + 16 + 8 * h);
        acc = __builtin_amdgcn_wmma_f32_16x16x32_bf16(false, a.v, false, b.v,
                                                      (short)0, acc, false, false);
    }
    int col = nt * 16 + m16;
    bool gate = dAct[col] > 1000;             // DEAD_AFTER
    #pragma unroll
    for (int r = 0; r < 8; ++r) {
        int row = mt * 16 + r + 8 * h;
        float v = gate ? acc[r] : 0.0f;
        dWb[(size_t)row * AUXK + col] = f2bf(v);
    }
}

// ============ 7. aux GEMM 2: aux_y + per-row aux squared error ============
__global__ __launch_bounds__(256) void k_aux_dec(const unsigned short* __restrict__ dWb,
                                                 const unsigned short* __restrict__ Bt,
                                                 const float* __restrict__ b_post,
                                                 const float* __restrict__ out,
                                                 float* __restrict__ auxSq) {
    int wave = threadIdx.x >> 5, lane = threadIdx.x & 31;
    int g = blockIdx.x * 8 + wave;            // 32768 tiles = (512 mt) x (64 nt)
    int mt = g >> 6, nt = g & 63;
    int m16 = lane & 15, h = lane >> 4;
    const unsigned short* aRow = dWb + (size_t)(mt * 16 + m16) * AUXK;
    const unsigned short* bRow = Bt + (size_t)(nt * 16 + m16) * AUXK;
    v8f acc = {0.f, 0.f, 0.f, 0.f, 0.f, 0.f, 0.f, 0.f};
    for (int kk = 0; kk < AUXK; kk += 32) {
        FragU a, b;
        a.q[0] = *(const uint4*)(aRow + kk + 8 * h);
        a.q[1] = *(const uint4*)(aRow + kk + 16 + 8 * h);
        b.q[0] = *(const uint4*)(bRow + kk + 8 * h);
        b.q[1] = *(const uint4*)(bRow + kk + 16 + 8 * h);
        acc = __builtin_amdgcn_wmma_f32_16x16x32_bf16(false, a.v, false, b.v,
                                                      (short)0, acc, false, false);
    }
    int col = nt * 16 + m16;
    float bp = b_post[col];
    #pragma unroll
    for (int r = 0; r < 8; ++r) {
        size_t row = (size_t)(mt * 16 + r + 8 * h);
        float v = acc[r] + bp;
        float d = out[OXN + row * DM + col] - v;
        float sq = d * d;
        sq += __shfl_xor(sq, 1, 32);
        sq += __shfl_xor(sq, 2, 32);
        sq += __shfl_xor(sq, 4, 32);
        sq += __shfl_xor(sq, 8, 32);
        if (m16 == 0) atomicAdd(&auxSq[row], sq);
    }
}

// ============ 8. per-column batch statistics ============
__global__ __launch_bounds__(256) void k_colstats(const float* __restrict__ out,
                                                  float* __restrict__ cs) {
    __shared__ float red[256];
    int d = blockIdx.x, tid = threadIdx.x;
    float sx = 0, sy = 0, sxx = 0, syy = 0, sxy = 0;
    for (int r = tid; r < BB; r += 256) {
        float x = out[OXN + (size_t)r * DM + d];
        float y = out[OYN + (size_t)r * DM + d];
        sx += x; sy += y; sxx += x * x; syy += y * y; sxy += x * y;
    }
    float tx = block_reduce(sx, red, tid, 256);
    float ty = block_reduce(sy, red, tid, 256);
    float txx = block_reduce(sxx, red, tid, 256);
    float tyy = block_reduce(syy, red, tid, 256);
    float txy = block_reduce(sxy, red, tid, 256);
    if (tid == 0) {
        cs[d * 5 + 0] = tx; cs[d * 5 + 1] = ty; cs[d * 5 + 2] = txx;
        cs[d * 5 + 3] = tyy; cs[d * 5 + 4] = txy;
    }
}

// ============ 9. finalize: aux_k_loss, loss, fvu, var_explained ============
__global__ __launch_bounds__(1024) void k_final(const float* __restrict__ auxSq,
                                                const float* __restrict__ cs,
                                                const int* __restrict__ n_steps,
                                                float* __restrict__ out) {
    __shared__ float red[1024];
    int tid = threadIdx.x;
    float gate = (n_steps[0] >= 1000) ? 1.0f : 0.0f;
    float sumRecon = 0.0f;
    for (int r = tid; r < BB; r += 1024) {
        float aux = gate * auxSq[r] * (1.0f / 1024.0f);
        out[OAL + r] = aux;
        float rl = out[ORL + r];
        out[OLS + r] = rl + 0.03125f * aux;
        sumRecon += rl;
    }
    float totRecon = block_reduce(sumRecon, red, tid, 1024);
    float sx = cs[tid * 5 + 0], sy = cs[tid * 5 + 1], sxx = cs[tid * 5 + 2];
    float syy = cs[tid * 5 + 3], sxy = cs[tid * 5 + 4];
    float mx = sx / (float)BB, my = sy / (float)BB;
    float cov = sxy / (float)BB - mx * my;
    float vx = sxx / (float)BB - mx * mx;
    float vy = syy / (float)BB - my * my;
    float ve = cov * cov / (vx * vy);
    float totVe = block_reduce(ve, red, tid, 1024);
    float totXX = block_reduce(sxx, red, tid, 1024);
    if (tid == 0) {
        float meanSq = totRecon / (float)BB;                 // mean((x-y)^2)
        float meanX2 = totXX / ((float)BB * (float)DM);      // mean(x^2)
        out[OFVU] = meanSq / meanX2;
        out[OVE] = totVe / (float)DM;
    }
}

// ============ host-side launcher ============
extern "C" void kernel_launch(void* const* d_in, const int* in_sizes, int n_in,
                              void* d_out, int out_size, void* d_ws, size_t ws_size,
                              hipStream_t stream) {
    const float* x        = (const float*)d_in[0];
    const float* W_enc    = (const float*)d_in[1];
    const float* W_dec    = (const float*)d_in[2];
    const float* b_pre    = (const float*)d_in[3];
    const float* b_post   = (const float*)d_in[4];
    const float* avg_norm = (const float*)d_in[5];
    const int*   act_in   = (const int*)d_in[6];
    const int*   n_steps  = (const int*)d_in[7];
    float* out = (float*)d_out;

    char* ws = (char*)d_ws;
    unsigned short* A    = (unsigned short*)(ws + WS_A);
    unsigned short* Wt   = (unsigned short*)(ws + WS_WT);
    int*            dIdx = (int*)(ws + WS_DIDX);
    int*            dAct = (int*)(ws + WS_DACT);
    unsigned short* Bd   = (unsigned short*)(ws + WS_BD);
    unsigned short* Bt   = (unsigned short*)(ws + WS_BT);
    unsigned short* dWb  = (unsigned short*)(ws + WS_DWB);
    float*          auxSq= (float*)(ws + WS_AUXSQ);
    float*          cs   = (float*)(ws + WS_CS);

    k_prep<<<(BB * DM) / 256, 256, 0, stream>>>(x, b_pre, b_post, avg_norm, out, A);
    k_wenc_t<<<dim3(NF / 32, DM / 32), dim3(32, 8), 0, stream>>>(W_enc, Wt);
    k_dead_topk<<<1, 1024, 0, stream>>>(act_in, dIdx, dAct);
    k_pack_enc<<<AUXK, 256, 0, stream>>>(W_enc, dIdx, Bd);
    k_pack_dec<<<AUXK, 256, 0, stream>>>(W_dec, dIdx, Bt);
    k_zero<<<(BB + 1023) / 1024, 1024, 0, stream>>>(auxSq, BB);

    k_main<<<BB / 64, 128, 0, stream>>>(A, Wt, W_dec, b_post, avg_norm, out);

    k_aux_enc<<<(BB / 16) * (AUXK / 16) / 8, 256, 0, stream>>>(A, Bd, dAct, dWb);
    k_aux_dec<<<(BB / 16) * (DM / 16) / 8, 256, 0, stream>>>(dWb, Bt, b_post, out, auxSq);

    k_colstats<<<DM, 256, 0, stream>>>(out, cs);
    k_final<<<1, 1024, 0, stream>>>(auxSq, cs, n_steps, out);
}
